// VQ_VAE_73349451481189
// MI455X (gfx1250) — compile-verified, matrix operand order checked by reading, not run on hardware
//
#include <hip/hip_runtime.h>

// ---------------------------------------------------------------------------
// VQ-VAE loss on MI455X (gfx1250, wave32, WMMA).
// loss = 2.25 * sum_rows( min_k ||x - e_k||^2 ) / (N*D)
// Cross term via v_wmma_f32_16x16x32_f16; norms exact in fp32.
// 4 passes x 8 accumulator tiles keeps the live set < 256 VGPRs so the
// compiler emits back-to-back WMMAs with no vgpr-msb window shuffling.
// ---------------------------------------------------------------------------

typedef _Float16 f16;
typedef __attribute__((ext_vector_type(8)))  f16   f16x8;
typedef __attribute__((ext_vector_type(16))) f16   v16h;
typedef __attribute__((ext_vector_type(8)))  float v8f;

#define D_DIM        480
#define K_CODES      512
#define N_ROWS       (64 * 4096)          // 262144
#define ROWS_PER_WG  128
#define WAVES_PER_WG 8
#define NUM_WG       (N_ROWS / ROWS_PER_WG)   // 2048
#define D_CHUNKS     (D_DIM / 32)             // 15 WMMA K-steps
#define N_PASSES     4                        // code passes
#define TILES_PP     8                        // 16-wide n-tiles per pass

// LDS row stride in f16 elements, padded so the 16 lanes of a half-wave hit
// distinct bank groups on ds_load_b128 (488*2B = 244 dwords, 244%64=52,
// gcd(52,64)=4 -> 16 distinct offsets for lanes 0..15).
#define LDS_ROW_STRIDE 488
#define XF16_BYTES  (ROWS_PER_WG * LDS_ROW_STRIDE * 2)          // 124928
#define SMEM_BYTES  (XF16_BYTES + ROWS_PER_WG * 4 + WAVES_PER_WG * 4)

// workspace layout (bytes)
#define WS_EF16_BYTES (K_CODES * (size_t)D_DIM * 2)   // 491520: f16 codebook
#define WS_ENORM_OFF  WS_EF16_BYTES                   // 512 floats: ||e_k||^2
#define WS_PART_OFF   (WS_ENORM_OFF + K_CODES * 4)    // 2048 floats: wg partials

// ---------------------------------------------------------------------------
// Kernel 1: codebook fp32 -> f16, plus exact fp32 squared norms.
// ---------------------------------------------------------------------------
__global__ void vq_prep(const float* __restrict__ emb,
                        f16* __restrict__ ef16,
                        float* __restrict__ enorm) {
    __shared__ float red[256];
    const int k = blockIdx.x;
    const int t = threadIdx.x;
    const float* row = emb + (size_t)k * D_DIM;
    f16* orow = ef16 + (size_t)k * D_DIM;
    float p = 0.0f;
    for (int idx = t; idx < D_DIM; idx += 256) {
        float x = row[idx];
        orow[idx] = (f16)x;
        p += x * x;
    }
    red[t] = p;
    __syncthreads();
    for (int s = 128; s > 0; s >>= 1) {
        if (t < s) red[t] += red[t + s];
        __syncthreads();
    }
    if (t == 0) enorm[k] = red[0];
}

// ---------------------------------------------------------------------------
// Kernel 2: per-row min distance over all 512 codes.
// Each wave owns 16 rows; workgroup = 8 waves = 128 rows. X is read from HBM
// exactly once (coalesced), staged as f16 in LDS; the 480 KB f16 codebook
// stays hot in the 192 MB L2 across all 2048 workgroups.
// ---------------------------------------------------------------------------
__launch_bounds__(256, 1)
__global__ void vq_main(const float* __restrict__ X,
                        const f16* __restrict__ ef16,
                        const float* __restrict__ enorm,
                        float* __restrict__ partials) {
    extern __shared__ char smem[];
    f16*   xf16  = (f16*)smem;
    float* xnorm = (float*)(smem + XF16_BYTES);
    float* wsum  = (float*)(smem + XF16_BYTES + ROWS_PER_WG * 4);

    const int tid  = threadIdx.x;
    const int lane = tid & 31;
    const int w    = tid >> 5;          // wave id 0..7
    const int mrow = lane & 15;
    const int hlf  = lane >> 4;         // 0 / 1
    const int koff = hlf * 8;           // 16-bit A/B per-lane K packing

    const int row0 = blockIdx.x * ROWS_PER_WG + w * 16;

    // ---- stage: coalesced fp32 load -> f16 in LDS + exact fp32 row norms ----
    for (int r = 0; r < 16; ++r) {
        const float* src = X + (size_t)(row0 + r) * D_DIM;
        f16* dst = xf16 + (size_t)(w * 16 + r) * LDS_ROW_STRIDE;
        float s = 0.0f;
        #pragma unroll
        for (int c = 0; c < D_CHUNKS; ++c) {
            float x = src[c * 32 + lane];      // 32 lanes -> 128B coalesced
            dst[c * 32 + lane] = (f16)x;
            s += x * x;
        }
        #pragma unroll
        for (int off = 16; off > 0; off >>= 1)
            s += __shfl_xor(s, off, 32);
        if (lane == 0) xnorm[w * 16 + r] = s;
    }
    __syncthreads();

    // A fragment base: 16-bit 16x32 A layout. Lane l holds row (l&15);
    // lanes 0-15 carry K {0..7,16..23}, lanes 16-31 carry K {8..15,24..31}.
    const f16* aBase = xf16 + (size_t)(w * 16 + mrow) * LDS_ROW_STRIDE + koff;

    float rm[8];
    #pragma unroll
    for (int g = 0; g < 8; ++g) rm[g] = 3.0e38f;

    // 512 codes in 4 passes of 8 n-tiles (64 accumulator VGPRs per pass:
    // whole live set stays in the directly-addressable VGPR 0..255 window).
    for (int jb = 0; jb < N_PASSES; ++jb) {
        v8f acc[TILES_PP] = {};
        for (int c = 0; c < D_CHUNKS; ++c) {
            const f16* ap = aBase + c * 32;
            f16x8 alo = *(const f16x8*)ap;          // ds_load_b128
            f16x8 ahi = *(const f16x8*)(ap + 16);   // ds_load_b128
            v16h a = __builtin_shufflevector(alo, ahi,
                0,1,2,3,4,5,6,7,8,9,10,11,12,13,14,15);
            #pragma unroll
            for (int j = 0; j < TILES_PP; ++j) {
                // B (32x16, K-contiguous in memory == codebook rows): mirrored
                // per-lane packing, lane index selects column n.
                const int n = (jb * TILES_PP + j) * 16 + mrow;
                const f16* bp = ef16 + (size_t)n * D_DIM + c * 32 + koff;
                f16x8 blo = *(const f16x8*)bp;      // global_load_b128 (L2-hot)
                f16x8 bhi = *(const f16x8*)(bp + 16);
                v16h b = __builtin_shufflevector(blo, bhi,
                    0,1,2,3,4,5,6,7,8,9,10,11,12,13,14,15);
                acc[j] = __builtin_amdgcn_wmma_f32_16x16x32_f16(
                    false, a, false, b, (short)0, acc[j], false, false);
            }
        }
        // fold ||e||^2 - 2*S and row-wise running min.
        // C layout: VGPR g, lanes 0-15 -> row g; lanes 16-31 -> row g+8.
        #pragma unroll
        for (int j = 0; j < TILES_PP; ++j) {
            float en = enorm[(jb * TILES_PP + j) * 16 + mrow];
            #pragma unroll
            for (int g = 0; g < 8; ++g)
                rm[g] = fminf(rm[g], en - 2.0f * acc[j][g]);
        }
    }

    // min across the 16 lanes of each half-wave (masks 1..8 keep halves apart)
    #pragma unroll
    for (int g = 0; g < 8; ++g) {
        #pragma unroll
        for (int m = 8; m > 0; m >>= 1)
            rm[g] = fminf(rm[g], __shfl_xor(rm[g], m, 32));
    }

    // add exact ||x||^2 once per row; lanes 0 and 16 own rows g and g+8.
    float contrib = 0.0f;
    if (mrow == 0) {
        #pragma unroll
        for (int g = 0; g < 8; ++g) {
            int r = hlf ? (g + 8) : g;
            contrib += rm[g] + xnorm[w * 16 + r];
        }
    }
    #pragma unroll
    for (int off = 16; off > 0; off >>= 1)
        contrib += __shfl_xor(contrib, off, 32);
    if (lane == 0) wsum[w] = contrib;
    __syncthreads();

    if (tid == 0) {  // fixed-order: deterministic
        float s = 0.0f;
        for (int i = 0; i < WAVES_PER_WG; ++i) s += wsum[i];
        partials[blockIdx.x] = s;
    }
}

// ---------------------------------------------------------------------------
// Kernel 3: deterministic fixed-order final reduction (double accumulation).
// ---------------------------------------------------------------------------
__global__ void vq_finish(const float* __restrict__ partials,
                          float* __restrict__ out) {
    __shared__ double red[256];
    const int t = threadIdx.x;
    double s = 0.0;
    for (int i = t; i < NUM_WG; i += 256) s += (double)partials[i];
    red[t] = s;
    __syncthreads();
    for (int k = 128; k > 0; k >>= 1) {
        if (t < k) red[t] += red[t + k];
        __syncthreads();
    }
    if (t == 0)
        out[0] = (float)(red[0] * (2.25 / ((double)N_ROWS * (double)D_DIM)));
}

// ---------------------------------------------------------------------------
extern "C" void kernel_launch(void* const* d_in, const int* in_sizes, int n_in,
                              void* d_out, int out_size, void* d_ws, size_t ws_size,
                              hipStream_t stream) {
    const float* X = (const float*)d_in[0];   // [64,4096,480] fp32
    const float* E = (const float*)d_in[1];   // [512,480] fp32
    char* ws = (char*)d_ws;
    f16*   ef16  = (f16*)ws;
    float* enorm = (float*)(ws + WS_ENORM_OFF);
    float* parts = (float*)(ws + WS_PART_OFF);

    vq_prep<<<K_CODES, 256, 0, stream>>>(E, ef16, enorm);
    vq_main<<<NUM_WG, 256, SMEM_BYTES, stream>>>(X, ef16, enorm, parts);
    vq_finish<<<1, 256, 0, stream>>>(parts, (float*)d_out);
}
